// AttentionModel_36885179138589
// MI455X (gfx1250) — matrix operand order repaired and checked by reading, not compiled, not run on hardware
//
#include <hip/hip_runtime.h>
#include <hip/hip_bf16.h>

// ---- Problem constants (from reference) ----
#define A_   8
#define NB   32768
#define D_   128
#define H_   4
#define K_   32
#define KP   33                    // padded LDS row stride (bank-conflict relief)
#define MATSZ (A_ * 16 * KP)       // one projection matrix tile in LDS (floats)

typedef __attribute__((ext_vector_type(16))) __bf16 v16bf;
typedef __attribute__((ext_vector_type(8)))  float  v8f;

// A-matrix bf16 16x32 fragment (this code's contiguous K order, not the ISA table order):
// lanes hold 16 contiguous K = base..base+15, base = half*16 (matches B fragment).
// Caller passes p = row_base + half*16 (+ t*32); first 8 floats at p, next 8 at p+8.
__device__ __forceinline__ v16bf make_a_frag(const float* __restrict__ p) {
    const float4* q0 = reinterpret_cast<const float4*>(p);
    const float4* q1 = reinterpret_cast<const float4*>(p + 8);
    float4 x0 = q0[0], x1 = q0[1], x2 = q1[0], x3 = q1[1];
    v16bf f;
    f[0]=(__bf16)x0.x; f[1]=(__bf16)x0.y; f[2]=(__bf16)x0.z; f[3]=(__bf16)x0.w;
    f[4]=(__bf16)x1.x; f[5]=(__bf16)x1.y; f[6]=(__bf16)x1.z; f[7]=(__bf16)x1.w;
    f[8]=(__bf16)x2.x; f[9]=(__bf16)x2.y; f[10]=(__bf16)x2.z; f[11]=(__bf16)x2.w;
    f[12]=(__bf16)x3.x; f[13]=(__bf16)x3.y; f[14]=(__bf16)x3.z; f[15]=(__bf16)x3.w;
    return f;
}

// B-matrix bf16 32x16 fragment: lane holds 16 contiguous K values
// (lanes 0-15: K=0..15; lanes 16-31: K=16..31) for its column (lane&15).
__device__ __forceinline__ v16bf make_b_frag(const float* __restrict__ p) {
    const float4* q = reinterpret_cast<const float4*>(p);
    float4 x0 = q[0], x1 = q[1], x2 = q[2], x3 = q[3];
    v16bf f;
    f[0]=(__bf16)x0.x; f[1]=(__bf16)x0.y; f[2]=(__bf16)x0.z; f[3]=(__bf16)x0.w;
    f[4]=(__bf16)x1.x; f[5]=(__bf16)x1.y; f[6]=(__bf16)x1.z; f[7]=(__bf16)x1.w;
    f[8]=(__bf16)x2.x; f[9]=(__bf16)x2.y; f[10]=(__bf16)x2.z; f[11]=(__bf16)x2.w;
    f[12]=(__bf16)x3.x; f[13]=(__bf16)x3.y; f[14]=(__bf16)x3.z; f[15]=(__bf16)x3.w;
    return f;
}

__device__ __forceinline__ v8f wmma_bf16(v16bf a, v16bf b, v8f c) {
    return __builtin_amdgcn_wmma_f32_16x16x32_bf16(
        /*neg_a=*/false, a, /*neg_b=*/false, b,
        /*c_mod=*/(short)0, c, /*reuse_a=*/false, /*reuse_b=*/false);
}

__global__ __launch_bounds__(256)
void agent_attn_fused(const float* __restrict__ querys,
                      const float* __restrict__ keys,
                      const float* __restrict__ values,
                      const float* __restrict__ Wk,
                      const float* __restrict__ Wsel,
                      const float* __restrict__ Wv,
                      const float* __restrict__ bv,
                      float* __restrict__ out)
{
    __shared__ float lds[3 * MATSZ];   // [mat 0=hk,1=hs,2=hv][agent][n_local][KP]

    const int tid  = threadIdx.x;
    const int w    = tid >> 5;         // wave id 0..7
    const int lane = tid & 31;
    const int half = lane >> 4;        // which 16-lane half
    const int arow = lane & 15;        // A-matrix row (n_local) / B column
    const int n0   = blockIdx.x * 16;  // batch tile base

    // ---- Hoist A fragments (Q/K/V rows as bf16 WMMA fragments) across heads ----
    // 24 tasks = 8 agents x {hk,hs,hv}; 3 per wave. Inputs total 384 MiB (3 x 8x32768x128 fp32),
    // fragments stay in VGPRs (3 tasks x 4 k-steps x 8 VGPRs = 96 VGPRs).
    int agents[3], mats[3];
    v16bf afrag[3][4];
    #pragma unroll
    for (int s = 0; s < 3; ++s) {
        int task = s * 8 + w;                 // 0..23
        int a = task / 3, m = task - a * 3;   // agent, matrix
        agents[s] = a; mats[s] = m;
        const float* src = (m == 0) ? keys : ((m == 1) ? querys : values);
        const float* base = src + ((size_t)a * NB + (size_t)(n0 + arow)) * D_ + half * 16;
        #pragma unroll
        for (int t = 0; t < 4; ++t)
            afrag[s][t] = make_a_frag(base + t * 32);
    }

    for (int h = 0; h < H_; ++h) {
        // ================= Phase 1: projections via WMMA =================
        #pragma unroll
        for (int s = 0; s < 3; ++s) {
            int a = agents[s], m = mats[s];
            const float* W  = (m == 0) ? Wk : ((m == 1) ? Wsel : Wv);
            const float* wb = W + (size_t)h * K_ * D_ + half * 16;
            v8f acc0 = {0.f,0.f,0.f,0.f,0.f,0.f,0.f,0.f};
            v8f acc1 = acc0;
            #pragma unroll
            for (int t = 0; t < 4; ++t) {     // K-dim: D=128 in 4 steps of 32
                v16bf b0 = make_b_frag(wb + (size_t)arow * D_        + t * 32); // feat 0..15
                v16bf b1 = make_b_frag(wb + (size_t)(16 + arow) * D_ + t * 32); // feat 16..31
                acc0 = wmma_bf16(afrag[s][t], b0, acc0);
                acc1 = wmma_bf16(afrag[s][t], b1, acc1);
            }
            // Scatter D fragment (f32 16x16 layout: VGPR v -> M = v + half*8, N = arow)
            float* dst = lds + m * MATSZ + a * (16 * KP);
            const int c0 = arow, c1 = 16 + arow;
            float bias0 = 0.f, bias1 = 0.f;
            if (m == 2) { bias0 = bv[h * K_ + c0]; bias1 = bv[h * K_ + c1]; }
            #pragma unroll
            for (int v = 0; v < 8; ++v) {
                int row = v + half * 8;
                float x0 = acc0[v], x1 = acc1[v];
                if (m == 2) {                 // hv: bias + leaky_relu(0.01)
                    x0 += bias0; x1 += bias1;
                    x0 = x0 > 0.f ? x0 : 0.01f * x0;
                    x1 = x1 > 0.f ? x1 : 0.01f * x1;
                }
                dst[row * KP + c0] = x0;
                dst[row * KP + c1] = x1;
            }
        }
        __syncthreads();

        // ================= Phase 2: logits / softmax / attend =================
        if (tid < 128) {
            const int nl = tid >> 3;          // n_local 0..15
            const int i  = tid & 7;           // query agent
            const float* hkB = lds + 0 * MATSZ;
            const float* hsB = lds + 1 * MATSZ;
            const float* hvB = lds + 2 * MATSZ;
            const float* hs  = hsB + i * (16 * KP) + nl * KP;

            float logit[8];
            #pragma unroll
            for (int j = 0; j < 8; ++j) {
                const float* hk = hkB + j * (16 * KP) + nl * KP;
                float d = 0.f;
                #pragma unroll
                for (int k = 0; k < K_; ++k) d += hs[k] * hk[k];
                logit[j] = d;                 // raw (unscaled) logits, as in reference
            }
            const float scale = 0.1767766952966369f;   // 1/sqrt(32)
            float mx = -3.0e38f;
            #pragma unroll
            for (int j = 0; j < 8; ++j)
                if (j != i) { float v = logit[j] * scale; mx = v > mx ? v : mx; }
            float e[8], sum = 0.f;
            #pragma unroll
            for (int j = 0; j < 8; ++j) {
                e[j] = (j == i) ? 0.f : __expf(logit[j] * scale - mx);
                sum += e[j];
            }
            const float inv = 1.f / sum;

            float acc[K_];
            #pragma unroll
            for (int k = 0; k < K_; ++k) acc[k] = 0.f;
            #pragma unroll
            for (int j = 0; j < 8; ++j) {
                float p = e[j] * inv;
                const float* hv = hvB + j * (16 * KP) + nl * KP;
                #pragma unroll
                for (int k = 0; k < K_; ++k) acc[k] += p * hv[k];
            }

            // ---- Outputs: other [A,H,N,32] ++ logits_ex [A,H,N,7] ++ probs_ex [A,H,N,7]
            const size_t n  = (size_t)n0 + nl;
            const size_t rb = ((size_t)i * H_ + h) * (size_t)NB + n;
            float* op = out + rb * K_;
            #pragma unroll
            for (int q = 0; q < 8; ++q)       // 128-bit coalesced stores
                reinterpret_cast<float4*>(op)[q] =
                    make_float4(acc[4*q], acc[4*q+1], acc[4*q+2], acc[4*q+3]);

            const size_t OFF1 = (size_t)A_ * H_ * NB * K_;
            const size_t OFF2 = OFF1 + (size_t)A_ * H_ * NB * 7;
            float* lp = out + OFF1 + rb * 7;
            float* pp = out + OFF2 + rb * 7;
            #pragma unroll
            for (int j = 0; j < 8; ++j) {
                if (j == i) continue;
                int jj = j - (j > i ? 1 : 0);
                lp[jj] = logit[j];
                pp[jj] = e[j] * inv;
            }
        }
        __syncthreads();
    }
}

extern "C" void kernel_launch(void* const* d_in, const int* in_sizes, int n_in,
                              void* d_out, int out_size, void* d_ws, size_t ws_size,
                              hipStream_t stream) {
    (void)in_sizes; (void)n_in; (void)out_size; (void)d_ws; (void)ws_size;
    const float* querys = (const float*)d_in[0];
    const float* keys   = (const float*)d_in[1];
    const float* values = (const float*)d_in[2];
    const float* Wk     = (const float*)d_in[3];
    const float* Wsel   = (const float*)d_in[4];
    const float* Wv     = (const float*)d_in[5];
    const float* bv     = (const float*)d_in[6];
    float* out = (float*)d_out;

    dim3 grid(NB / 16);   // 2048 blocks, each owns 16 batch rows for all agents/heads
    dim3 block(256);      // 8 wave32s
    agent_attn_fused<<<grid, block, 0, stream>>>(querys, keys, values,
                                                 Wk, Wsel, Wv, bv, out);
}
